// BiLSTM_63110249447498
// MI455X (gfx1250) — compile-verified, matrix operand order checked by reading, not run on hardware
//
#include <hip/hip_runtime.h>
#include <hip/hip_bf16.h>

typedef __attribute__((ext_vector_type(16))) _Float16 v16h;
typedef __attribute__((ext_vector_type(8)))  _Float16 v8h;
typedef __attribute__((ext_vector_type(8)))  float    v8f;
typedef __attribute__((ext_vector_type(4)))  float    v4f;

#define NPOS 16384   // B*H*W = 16*32*32

__device__ __forceinline__ v8f zero8f() {
    v8f z = {0.f,0.f,0.f,0.f,0.f,0.f,0.f,0.f};
    return z;
}
__device__ __forceinline__ v16h zero16h() {
    v16h z = {(_Float16)0,(_Float16)0,(_Float16)0,(_Float16)0,
              (_Float16)0,(_Float16)0,(_Float16)0,(_Float16)0,
              (_Float16)0,(_Float16)0,(_Float16)0,(_Float16)0,
              (_Float16)0,(_Float16)0,(_Float16)0,(_Float16)0};
    return z;
}
// 16x16x32 f16 fragment per-lane data: lanes 0-15 hold K {0..7,16..23},
// lanes 16-31 hold K {8..15,24..31} of their row/col; p already includes half*8.
__device__ __forceinline__ v16h frag16(const _Float16* p) {
    v8h lo = *(const v8h*)(p);
    v8h hi = *(const v8h*)(p + 16);
    return __builtin_shufflevector(lo, hi, 0,1,2,3,4,5,6,7,8,9,10,11,12,13,14,15);
}
__device__ __forceinline__ v8f wmma_f16(v16h a, v16h b, v8f c) {
    return __builtin_amdgcn_wmma_f32_16x16x32_f16(false, a, false, b,
                                                  (short)0, c, false, false);
}
__device__ __forceinline__ float sigmoidf_(float v) {
    return 1.0f / (1.0f + __expf(-v));
}

// ---------------------------------------------------------------------------
// Prep: NCHW f32 -> [pos][128] f16; pack weights f16; zero LSTM states.
// ---------------------------------------------------------------------------
__global__ void prep_kernel(const float* __restrict__ x,
                            const float* __restrict__ wi,
                            const float* __restrict__ wl,
                            const float* __restrict__ wr,
                            const float* __restrict__ wsk,
                            _Float16* __restrict__ XT,
                            _Float16* __restrict__ Wi,
                            _Float16* __restrict__ Wl,
                            _Float16* __restrict__ Wr,
                            _Float16* __restrict__ Wsk,
                            _Float16* __restrict__ lh0,
                            _Float16* __restrict__ rh0,
                            float* __restrict__ lc,
                            float* __restrict__ rc)
{
    int tid = blockIdx.x * blockDim.x + threadIdx.x;
    int nth = gridDim.x * blockDim.x;
    // x: [16,128,32,32] -> XT [pos][128]
    for (int i = tid; i < NPOS * 128; i += nth) {
        int pos = i >> 7, c = i & 127;
        int b = pos >> 10, hw = pos & 1023;
        XT[i] = (_Float16)x[(((size_t)b * 128 + c) << 10) + hw];
    }
    // w_i2s [256][128] already (o,c) row-major
    for (int i = tid; i < 256 * 128; i += nth) Wi[i] = (_Float16)wi[i];
    // w_left/right [256][64][2] -> packed [o][k], k<64 -> tap0 (hd), k>=64 -> tap1 (hw)
    for (int i = tid; i < 256 * 128; i += nth) {
        int o = i >> 7, k = i & 127;
        int j = k >> 6, c = k & 63;
        Wl[i] = (_Float16)wl[o * 128 + c * 2 + j];
        Wr[i] = (_Float16)wr[o * 128 + c * 2 + j];
    }
    for (int i = tid; i < 128 * 64; i += nth) Wsk[i] = (_Float16)wsk[i];
    for (int i = tid; i < NPOS * 64; i += nth) {
        lh0[i] = (_Float16)0; rh0[i] = (_Float16)0;
        lc[i] = 0.f; rc[i] = 0.f;
    }
}

// ---------------------------------------------------------------------------
// hmap[pos][256] = XT[pos][128] . Wi[256][128]^T   (M=channel, N=pos tiles)
// ---------------------------------------------------------------------------
__global__ __launch_bounds__(256) void hmap_kernel(
    const _Float16* __restrict__ XT,
    const _Float16* __restrict__ Wi,
    float* __restrict__ hmap)
{
    int wave = (blockIdx.x * blockDim.x + threadIdx.x) >> 5;
    int lane = threadIdx.x & 31;
    int ntile = wave;                       // 0..1023
    int lp = lane & 15, half = lane >> 4;
    int pos = ntile * 16 + lp;

    const _Float16* brow = XT + (size_t)pos * 128;
    v16h bf0 = frag16(brow +  0 + half * 8);
    v16h bf1 = frag16(brow + 32 + half * 8);
    v16h bf2 = frag16(brow + 64 + half * 8);
    v16h bf3 = frag16(brow + 96 + half * 8);

    float* hrow = hmap + (size_t)pos * 256;
    for (int mt = 0; mt < 16; ++mt) {
        const _Float16* arow = Wi + (size_t)(mt * 16 + lp) * 128;
        v8f acc = zero8f();
        acc = wmma_f16(frag16(arow +  0 + half * 8), bf0, acc);
        acc = wmma_f16(frag16(arow + 32 + half * 8), bf1, acc);
        acc = wmma_f16(frag16(arow + 64 + half * 8), bf2, acc);
        acc = wmma_f16(frag16(arow + 96 + half * 8), bf3, acc);
        int ch0 = mt * 16 + half * 8;
        *(v4f*)(hrow + ch0)     = __builtin_shufflevector(acc, acc, 0,1,2,3);
        *(v4f*)(hrow + ch0 + 4) = __builtin_shufflevector(acc, acc, 4,5,6,7);
    }
}

// ---------------------------------------------------------------------------
// One scan step, both directions. dir=0 left (shift right/down), dir=1 right.
// pre = sigmoid(W0.hd + W1.hw + bias + hmap); gates o|fg|ig|g (64 ch each);
// c = fg*c + ig*g ; h = o*tanh(c)
// ---------------------------------------------------------------------------
__global__ __launch_bounds__(256) void step_kernel(
    const float* __restrict__ hmap,
    const _Float16* __restrict__ Wl, const _Float16* __restrict__ Wr,
    const float* __restrict__ bl,    const float* __restrict__ br,
    const _Float16* __restrict__ lh_prev, _Float16* __restrict__ lh_next,
    const _Float16* __restrict__ rh_prev, _Float16* __restrict__ rh_next,
    float* __restrict__ lc, float* __restrict__ rc)
{
    int wave = (blockIdx.x * blockDim.x + threadIdx.x) >> 5;
    int lane = threadIdx.x & 31;
    int dir   = wave & 1;
    int ntile = wave >> 1;                  // 0..1023
    int b   = ntile >> 6;
    int rem = ntile & 63;
    int h   = rem >> 1;
    int w0  = (rem & 1) << 4;
    int lp = lane & 15, half = lane >> 4;
    int w = w0 + lp;
    int pos = ((b << 5) + h) * 32 + w;

    const _Float16* Hp = dir ? rh_prev : lh_prev;
    _Float16*       Hn = dir ? rh_next : lh_next;
    float*          Cs = dir ? rc : lc;
    const _Float16* Wd = dir ? Wr : Wl;
    const float*    bias = dir ? br : bl;

    int wsrc = dir ? (w + 1) : (w - 1);
    bool okw = (unsigned)wsrc < 32u;
    bool okhd = okw && (h > 0);
    int posHW = ((b << 5) + h) * 32 + wsrc;
    int posHD = posHW - 32;

    v16h zf = zero16h();
    const _Float16* rowHD = Hp + (size_t)posHD * 64;
    const _Float16* rowHW = Hp + (size_t)posHW * 64;
    v16h bf0 = okhd ? frag16(rowHD +  0 + half * 8) : zf;  // hd, k 0..31
    v16h bf1 = okhd ? frag16(rowHD + 32 + half * 8) : zf;  // hd, k 32..63
    v16h bf2 = okw  ? frag16(rowHW +  0 + half * 8) : zf;  // hw, k 0..31
    v16h bf3 = okw  ? frag16(rowHW + 32 + half * 8) : zf;  // hw, k 32..63

    const float* hrow = hmap + (size_t)pos * 256;
    float*       crow = Cs   + (size_t)pos * 64;
    _Float16*    hout = Hn   + (size_t)pos * 64;

    for (int t = 0; t < 4; ++t) {
        v8f acc[4];                         // g=0:o  1:fg  2:ig  3:g
        #pragma unroll
        for (int g = 0; g < 4; ++g) {
            int mt = t + g * 4;
            const _Float16* arow = Wd + (size_t)(mt * 16 + lp) * 128;
            v8f c8 = zero8f();
            c8 = wmma_f16(frag16(arow +  0 + half * 8), bf0, c8);
            c8 = wmma_f16(frag16(arow + 32 + half * 8), bf1, c8);
            c8 = wmma_f16(frag16(arow + 64 + half * 8), bf2, c8);
            c8 = wmma_f16(frag16(arow + 96 + half * 8), bf3, c8);
            acc[g] = c8;
        }
        int fbase = t * 16 + half * 8;      // gate-local channel base (0..63)
        // hmap + bias slices for the four gate groups
        v4f hm[4][2], bi[4][2];
        #pragma unroll
        for (int g = 0; g < 4; ++g) {
            int ch0 = g * 64 + fbase;
            hm[g][0] = *(const v4f*)(hrow + ch0);
            hm[g][1] = *(const v4f*)(hrow + ch0 + 4);
            bi[g][0] = *(const v4f*)(bias + ch0);
            bi[g][1] = *(const v4f*)(bias + ch0 + 4);
        }
        v4f c_old0 = *(const v4f*)(crow + fbase);
        v4f c_old1 = *(const v4f*)(crow + fbase + 4);
        v4f c_new0, c_new1;
        v8h hvec = {(_Float16)0,(_Float16)0,(_Float16)0,(_Float16)0,
                    (_Float16)0,(_Float16)0,(_Float16)0,(_Float16)0};
        #pragma unroll
        for (int v = 0; v < 8; ++v) {
            int q = v >> 2, r = v & 3;
            float po  = sigmoidf_(acc[0][v] + hm[0][q][r] + bi[0][q][r]);
            float pfg = sigmoidf_(acc[1][v] + hm[1][q][r] + bi[1][q][r]);
            float pig = sigmoidf_(acc[2][v] + hm[2][q][r] + bi[2][q][r]);
            float pg  = sigmoidf_(acc[3][v] + hm[3][q][r] + bi[3][q][r]);
            float cold = (q == 0) ? c_old0[r] : c_old1[r];
            float cn = pfg * cold + pig * pg;
            if (q == 0) c_new0[r] = cn; else c_new1[r] = cn;
            hvec[v] = (_Float16)(po * tanhf(cn));
        }
        *(v4f*)(crow + fbase)     = c_new0;
        *(v4f*)(crow + fbase + 4) = c_new1;
        *(v8h*)(hout + fbase)     = hvec;
    }
}

// ---------------------------------------------------------------------------
// out = x + W_skip.(lh + shift_down(rh)) + b_skip, scattered back to NCHW f32
// ---------------------------------------------------------------------------
__global__ __launch_bounds__(256) void final_kernel(
    const _Float16* __restrict__ lh, const _Float16* __restrict__ rh,
    const _Float16* __restrict__ Wsk, const float* __restrict__ bsk,
    const float* __restrict__ x, float* __restrict__ out)
{
    int wave = (blockIdx.x * blockDim.x + threadIdx.x) >> 5;
    int lane = threadIdx.x & 31;
    int ntile = wave;
    int b   = ntile >> 6;
    int rem = ntile & 63;
    int h   = rem >> 1;
    int w0  = (rem & 1) << 4;
    int lp = lane & 15, half = lane >> 4;
    int w = w0 + lp;
    int pos = ((b << 5) + h) * 32 + w;

    const _Float16* lrow = lh + (size_t)pos * 64;
    const _Float16* rrow = rh + (size_t)(pos - 32) * 64;  // shift_down: rh[h-1]
    bool okh = (h > 0);

    v16h bf[2];
    #pragma unroll
    for (int c = 0; c < 2; ++c) {
        int p = c * 32 + half * 8;
        v8h a0 = *(const v8h*)(lrow + p);
        v8h a1 = *(const v8h*)(lrow + p + 16);
        if (okh) {
            a0 = a0 + *(const v8h*)(rrow + p);
            a1 = a1 + *(const v8h*)(rrow + p + 16);
        }
        bf[c] = __builtin_shufflevector(a0, a1, 0,1,2,3,4,5,6,7,
                                        8,9,10,11,12,13,14,15);
    }

    size_t sp = (size_t)h * 32 + w;         // spatial index within image
    for (int mt = 0; mt < 8; ++mt) {
        const _Float16* arow = Wsk + (size_t)(mt * 16 + lp) * 64;
        v8f acc = zero8f();
        acc = wmma_f16(frag16(arow +  0 + half * 8), bf[0], acc);
        acc = wmma_f16(frag16(arow + 32 + half * 8), bf[1], acc);
        int ch0 = mt * 16 + half * 8;
        v4f b0 = *(const v4f*)(bsk + ch0);
        v4f b1 = *(const v4f*)(bsk + ch0 + 4);
        #pragma unroll
        for (int v = 0; v < 8; ++v) {
            int ch = ch0 + v;
            size_t idx = (((size_t)b * 128 + ch) << 10) + sp;
            float bv = (v < 4) ? b0[v] : b1[v - 4];
            out[idx] = x[idx] + acc[v] + bv;
        }
    }
}

// ---------------------------------------------------------------------------
extern "C" void kernel_launch(void* const* d_in, const int* in_sizes, int n_in,
                              void* d_out, int out_size, void* d_ws, size_t ws_size,
                              hipStream_t stream) {
    const float* x       = (const float*)d_in[0];
    const float* w_i2s   = (const float*)d_in[1];
    const float* w_left  = (const float*)d_in[2];
    const float* b_left  = (const float*)d_in[3];
    const float* w_right = (const float*)d_in[4];
    const float* b_right = (const float*)d_in[5];
    const float* w_skip  = (const float*)d_in[6];
    const float* b_skip  = (const float*)d_in[7];
    float* out = (float*)d_out;

    char* ws = (char*)d_ws;
    float*    hmap = (float*)   (ws + 0);          // 16384*256 f32 = 16 MB
    _Float16* XT   = (_Float16*)(ws + 16777216);   // 16384*128 f16 =  4 MB
    _Float16* Wi   = (_Float16*)(ws + 20971520);   // 256*128 f16
    _Float16* Wl   = (_Float16*)(ws + 21037056);
    _Float16* Wr   = (_Float16*)(ws + 21102592);
    _Float16* Wsk  = (_Float16*)(ws + 21168128);   // 128*64 f16
    _Float16* lhb[2] = {(_Float16*)(ws + 21184512), (_Float16*)(ws + 23281664)};
    _Float16* rhb[2] = {(_Float16*)(ws + 25378816), (_Float16*)(ws + 27475968)};
    float* lc = (float*)(ws + 29573120);           // 16384*64 f32 = 4 MB
    float* rc = (float*)(ws + 33767424);

    prep_kernel<<<256, 256, 0, stream>>>(x, w_i2s, w_left, w_right, w_skip,
                                         XT, Wi, Wl, Wr, Wsk,
                                         lhb[0], rhb[0], lc, rc);
    hmap_kernel<<<128, 256, 0, stream>>>(XT, Wi, hmap);
    for (int s = 0; s < 32; ++s) {
        int cur = s & 1, nxt = cur ^ 1;
        step_kernel<<<256, 256, 0, stream>>>(hmap, Wl, Wr, b_left, b_right,
                                             lhb[cur], lhb[nxt],
                                             rhb[cur], rhb[nxt], lc, rc);
    }
    final_kernel<<<128, 256, 0, stream>>>(lhb[0], rhb[0], Wsk, b_skip, x, out);
}
